// MultiHeadAttention_22067541967542
// MI455X (gfx1250) — compile-verified, matrix operand order checked by reading, not run on hardware
//
#include <hip/hip_runtime.h>
#include <hip/hip_bf16.h>
#include <math.h>

// ---------------------------------------------------------------------------
// Types for CDNA5 WMMA (wave32): v_wmma_f32_16x16x32_bf16
// ---------------------------------------------------------------------------
typedef __bf16 bf16_t;
typedef __attribute__((ext_vector_type(16))) __bf16 v16bf;
typedef __attribute__((ext_vector_type(8)))  __bf16 v8bf;
typedef __attribute__((ext_vector_type(8)))  float  v8f;

#define WMMA_BF16(a, b, c) \
  __builtin_amdgcn_wmma_f32_16x16x32_bf16(false, (a), false, (b), (short)0, (c), false, false)

// A-fragment (16x32 bf16) per-lane layout: lane = 16*half + m.
// half 0 holds K {kk..kk+7, kk+16..kk+23}; half 1 holds K {kk+8..kk+15, kk+24..kk+31}.
// With row-major source (stride >= 32 along K) that is two 16-byte loads.
__device__ __forceinline__ v16bf load_frag_bf16(const bf16_t* p) {
  v8bf lo = *(const v8bf*)p;
  v8bf hi = *(const v8bf*)(p + 16);
  v16bf f;
#pragma unroll
  for (int i = 0; i < 8; ++i) { f[i] = lo[i]; f[i + 8] = hi[i]; }
  return f;
}

// ---------------------------------------------------------------------------
// Problem constants
// ---------------------------------------------------------------------------
#define BATCH 4
#define SEQ   1024
#define DIM   1024
#define HEADS 16
#define DH    64
#define BH    (BATCH * HEADS)      // 64
#define ROWS  (BATCH * SEQ)        // 4096

// ---------------------------------------------------------------------------
// 1) BatchNorm(q) -> bf16
// ---------------------------------------------------------------------------
__global__ void k_bn(const float* __restrict__ q, const float* __restrict__ gamma,
                     const float* __restrict__ beta, const float* __restrict__ mean,
                     const float* __restrict__ var, bf16_t* __restrict__ xb) {
  size_t idx = (size_t)blockIdx.x * blockDim.x + threadIdx.x;  // ROWS*DIM threads
  int d = (int)(idx & (DIM - 1));
  float x = (q[idx] - mean[d]) * rsqrtf(var[d] + 1e-3f) * gamma[d] + beta[d];
  xb[idx] = (bf16_t)x;
}

// ---------------------------------------------------------------------------
// 2) Weight fp32 [K x N] -> transposed bf16 [N x K] (so B-fragments are row loads)
// ---------------------------------------------------------------------------
__global__ void k_wT(const float* __restrict__ W, bf16_t* __restrict__ WT) {
  size_t idx = (size_t)blockIdx.x * blockDim.x + threadIdx.x;  // DIM*DIM threads
  int k = (int)(idx >> 10);
  int n = (int)(idx & (DIM - 1));
  WT[(size_t)n * DIM + k] = (bf16_t)W[idx];
}

// ---------------------------------------------------------------------------
// 3) Generic bf16 WMMA GEMM: C[M x N] = epilogue(A[M x K] @ BT[N x K]^T + bias)
//    Wave tile 16x64 (4 accumulators, A reused 4x); block = 8 waves = 128 rows.
//    MODE 0: relu -> bf16 row-major          (GEMM1)
//    MODE 1: split-heads scatter -> qh [b,h,s,dh] and qh^T [b,h,dh,s]  (GEMM2)
//    MODE 2: fp32 row-major store            (GEMM3 -> d_out)
// ---------------------------------------------------------------------------
template <int MODE>
__global__ void k_gemm(const bf16_t* __restrict__ A, const bf16_t* __restrict__ BT,
                       const float* __restrict__ bias, bf16_t* __restrict__ out_b,
                       float* __restrict__ out_f, bf16_t* __restrict__ out_t,
                       int M, int N, int K) {
  const int wave = threadIdx.x >> 5;
  const int lane = threadIdx.x & 31;
  const int half = lane >> 4;
  const int lrow = lane & 15;
  const int m0 = blockIdx.x * 128 + wave * 16;
  const int n0 = blockIdx.y * 64;

  v8f acc[4] = {};
  const bf16_t* Ap = A + (size_t)(m0 + lrow) * K + half * 8;
#pragma unroll 4
  for (int kk = 0; kk < K; kk += 32) {
    v16bf a = load_frag_bf16(Ap + kk);
#pragma unroll
    for (int j = 0; j < 4; ++j) {
      const bf16_t* Bp = BT + (size_t)(n0 + j * 16 + lrow) * K + kk + half * 8;
      v16bf b = load_frag_bf16(Bp);
      acc[j] = WMMA_BF16(a, b, acc[j]);
    }
  }

  const int mbase = m0 + half * 8;
#pragma unroll
  for (int j = 0; j < 4; ++j) {
    const int n = n0 + j * 16 + lrow;
    const float bv = bias[n];
#pragma unroll
    for (int r = 0; r < 8; ++r) {
      const int m = mbase + r;
      float val = acc[j][r] + bv;
      if constexpr (MODE == 0) {
        val = fmaxf(val, 0.0f);
        out_b[(size_t)m * N + n] = (bf16_t)val;
      } else if constexpr (MODE == 1) {
        const int b_ = m >> 10, s = m & (SEQ - 1);
        const int h = n >> 6, dl = n & (DH - 1);
        const size_t bh = (size_t)b_ * HEADS + h;
        out_b[(bh * SEQ + s) * DH + dl] = (bf16_t)val;          // qh [b,h,s,dh]
        out_t[(bh * DH + dl) * SEQ + s] = (bf16_t)val;          // qh^T [b,h,dh,s]
      } else {
        out_f[(size_t)m * N + n] = val;
      }
    }
  }
}

// ---------------------------------------------------------------------------
// 4) logits[bh] = qh[bh] @ qh[bh]^T * (1/8) + mask * (-1e9)  (K = dh = 64)
//    stored fp32 straight into the attn region of d_out.
// ---------------------------------------------------------------------------
__global__ void k_logits(const bf16_t* __restrict__ qh, const int* __restrict__ mask,
                         float* __restrict__ logits) {
  const int bh = blockIdx.z;
  const int b = bh >> 4;
  const int wave = threadIdx.x >> 5;
  const int lane = threadIdx.x & 31;
  const int half = lane >> 4;
  const int lrow = lane & 15;
  const int m0 = blockIdx.x * 128 + wave * 16;
  const int n0 = blockIdx.y * 64;

  const bf16_t* Ab = qh + (size_t)bh * SEQ * DH;
  v8f acc[4] = {};
  const bf16_t* Ap = Ab + (size_t)(m0 + lrow) * DH + half * 8;
#pragma unroll
  for (int kk = 0; kk < DH; kk += 32) {
    v16bf a = load_frag_bf16(Ap + kk);
#pragma unroll
    for (int j = 0; j < 4; ++j) {
      v16bf bfr = load_frag_bf16(Ab + (size_t)(n0 + j * 16 + lrow) * DH + kk + half * 8);
      acc[j] = WMMA_BF16(a, bfr, acc[j]);
    }
  }

  const int mbase = m0 + half * 8;
  const int* mb = mask + (size_t)b * SEQ * SEQ;
  float* lg = logits + (size_t)bh * SEQ * SEQ;
#pragma unroll
  for (int j = 0; j < 4; ++j) {
    const int n = n0 + j * 16 + lrow;
#pragma unroll
    for (int r = 0; r < 8; ++r) {
      const int m = mbase + r;
      const float mv = (float)mb[(size_t)m * SEQ + n];
      lg[(size_t)m * SEQ + n] = acc[j][r] * 0.125f + mv * -1e9f;
    }
  }
}

// ---------------------------------------------------------------------------
// 5) Per-row (bh*SEQ rows): max, sum(exp(x-max)), mean — single read pass.
// ---------------------------------------------------------------------------
__global__ void k_rowstats(const float* __restrict__ logits, float* __restrict__ rowmax,
                           float* __restrict__ rowsumexp, float* __restrict__ rowmean) {
  const int row = blockIdx.x;
  const int tid = threadIdx.x;
  const float* p = logits + (size_t)row * SEQ;
  float v[4], lmax = -3.402823466e38f, lsum = 0.0f;
#pragma unroll
  for (int i = 0; i < 4; ++i) {
    v[i] = p[tid + i * 256];
    lmax = fmaxf(lmax, v[i]);
    lsum += v[i];
  }
  __shared__ float smax[256], ssum[256];
  smax[tid] = lmax; ssum[tid] = lsum;
  __syncthreads();
  for (int s = 128; s > 0; s >>= 1) {
    if (tid < s) { smax[tid] = fmaxf(smax[tid], smax[tid + s]); ssum[tid] += ssum[tid + s]; }
    __syncthreads();
  }
  const float gmax = smax[0];
  const float gsum = ssum[0];
  __syncthreads();
  float lse = 0.0f;
#pragma unroll
  for (int i = 0; i < 4; ++i) lse += __expf(v[i] - gmax);
  ssum[tid] = lse;
  __syncthreads();
  for (int s = 128; s > 0; s >>= 1) {
    if (tid < s) ssum[tid] += ssum[tid + s];
    __syncthreads();
  }
  if (tid == 0) {
    rowmax[row] = gmax;
    rowsumexp[row] = ssum[0];
    rowmean[row] = gsum * (1.0f / SEQ);
  }
}

// ---------------------------------------------------------------------------
// 6) w_within[bh, q] = softmax_q( znorm(rowmean over q, scale 2) ); last2[bh] = mu
// ---------------------------------------------------------------------------
__global__ void k_within(const float* __restrict__ rowmean, float* __restrict__ w_within,
                         float* __restrict__ last2) {
  const int bh = blockIdx.x;
  const int tid = threadIdx.x;
  const float* p = rowmean + (size_t)bh * SEQ;
  float v[4], s = 0.0f, s2 = 0.0f;
#pragma unroll
  for (int i = 0; i < 4; ++i) { v[i] = p[tid + i * 256]; s += v[i]; s2 += v[i] * v[i]; }
  __shared__ float sa[256], sb[256];
  sa[tid] = s; sb[tid] = s2;
  __syncthreads();
  for (int st = 128; st > 0; st >>= 1) {
    if (tid < st) { sa[tid] += sa[tid + st]; sb[tid] += sb[tid + st]; }
    __syncthreads();
  }
  const float mu = sa[0] * (1.0f / SEQ);
  const float var = sb[0] * (1.0f / SEQ) - mu * mu;
  const float sd = sqrtf(fmaxf(var, 0.0f));
  const float inv = 1.0f / (sd * 2.0f + 1e-10f);
  __syncthreads();
  float t[4], lmax = -3.402823466e38f;
#pragma unroll
  for (int i = 0; i < 4; ++i) { t[i] = (v[i] - mu) * inv; lmax = fmaxf(lmax, t[i]); }
  sa[tid] = lmax;
  __syncthreads();
  for (int st = 128; st > 0; st >>= 1) {
    if (tid < st) sa[tid] = fmaxf(sa[tid], sa[tid + st]);
    __syncthreads();
  }
  const float tmax = sa[0];
  __syncthreads();
  float ls = 0.0f;
#pragma unroll
  for (int i = 0; i < 4; ++i) { t[i] = __expf(t[i] - tmax); ls += t[i]; }
  sb[tid] = ls;
  __syncthreads();
  for (int st = 128; st > 0; st >>= 1) {
    if (tid < st) sb[tid] += sb[tid + st];
    __syncthreads();
  }
  const float invsum = 1.0f / sb[0];
#pragma unroll
  for (int i = 0; i < 4; ++i) w_within[(size_t)bh * SEQ + tid + i * 256] = t[i] * invsum;
  if (tid == 0) last2[bh] = mu;
}

// ---------------------------------------------------------------------------
// 7) w_betw[b, h] = softmax_h( znorm(last2 over heads, scale 2) )
// ---------------------------------------------------------------------------
__global__ void k_betw(const float* __restrict__ last2, float* __restrict__ w_betw) {
  const int b = threadIdx.x;
  if (b >= BATCH) return;
  float x[HEADS], mu = 0.0f;
#pragma unroll
  for (int h = 0; h < HEADS; ++h) { x[h] = last2[b * HEADS + h]; mu += x[h]; }
  mu *= (1.0f / HEADS);
  float var = 0.0f;
#pragma unroll
  for (int h = 0; h < HEADS; ++h) { float d = x[h] - mu; var += d * d; }
  var *= (1.0f / HEADS);
  const float inv = 1.0f / (sqrtf(var) * 2.0f + 1e-10f);
  float tmax = -3.402823466e38f;
#pragma unroll
  for (int h = 0; h < HEADS; ++h) { x[h] = (x[h] - mu) * inv; tmax = fmaxf(tmax, x[h]); }
  float sum = 0.0f;
#pragma unroll
  for (int h = 0; h < HEADS; ++h) { x[h] = __expf(x[h] - tmax); sum += x[h]; }
  const float invsum = 1.0f / sum;
#pragma unroll
  for (int h = 0; h < HEADS; ++h) w_betw[b * HEADS + h] = x[h] * invsum;
}

// ---------------------------------------------------------------------------
// 8) Fused finalize + attn@qh:
//    Reads raw logits from the attn region, applies
//      attn = exp(logit - rowmax) * (w_within * w_betw / rowsumexp)
//    in-register, writes finalized attn back (same addresses, disjoint per-lane
//    ownership -> no races), converts to bf16 and runs the [S x S] @ [S x dh]
//    WMMA GEMM. Output merged-heads bf16 [B,S,D]. Saves a full 268 MB pass.
// ---------------------------------------------------------------------------
__global__ void k_attn_out_fused(float* __restrict__ attn, const bf16_t* __restrict__ qh_t,
                                 const float* __restrict__ rowmax,
                                 const float* __restrict__ rowsumexp,
                                 const float* __restrict__ w_within,
                                 const float* __restrict__ w_betw,
                                 bf16_t* __restrict__ outb) {
  const int bh = blockIdx.z;
  const int b = bh >> 4, h = bh & 15;
  const int wave = threadIdx.x >> 5;
  const int lane = threadIdx.x & 31;
  const int half = lane >> 4;
  const int lrow = lane & 15;
  const int m0 = blockIdx.x * 128 + wave * 16;
  const int m = m0 + lrow;                          // row this lane owns
  const size_t rowid = (size_t)bh * SEQ + m;

  const float rmax = rowmax[rowid];
  const float scale = w_within[rowid] * w_betw[bh] / rowsumexp[rowid];

  float* Ap = attn + rowid * SEQ + half * 8;        // lane's k-slice of the row
  const bf16_t* Bb = qh_t + (size_t)bh * DH * SEQ;
  v8f acc[4] = {};
#pragma unroll 2
  for (int kk = 0; kk < SEQ; kk += 32) {
    v8f lo = *(v8f*)(Ap + kk);
    v8f hi = *(v8f*)(Ap + kk + 16);
    v16bf a;
#pragma unroll
    for (int i = 0; i < 8; ++i) {
      const float e0 = __expf(lo[i] - rmax) * scale;
      const float e1 = __expf(hi[i] - rmax) * scale;
      lo[i] = e0; hi[i] = e1;
      a[i] = (bf16_t)e0;
      a[i + 8] = (bf16_t)e1;
    }
    *(v8f*)(Ap + kk) = lo;                          // finalized attn out
    *(v8f*)(Ap + kk + 16) = hi;
#pragma unroll
    for (int j = 0; j < 4; ++j) {
      v16bf bfr = load_frag_bf16(Bb + (size_t)(j * 16 + lrow) * SEQ + kk + half * 8);
      acc[j] = WMMA_BF16(a, bfr, acc[j]);
    }
  }
  const int mbase = m0 + half * 8;
#pragma unroll
  for (int j = 0; j < 4; ++j) {
    const int n = j * 16 + lrow;
#pragma unroll
    for (int r = 0; r < 8; ++r) {
      const int mm = mbase + r;
      outb[((size_t)b * SEQ + mm) * DIM + h * DH + n] = (bf16_t)acc[j][r];
    }
  }
}

// ---------------------------------------------------------------------------
// Host launcher
// ---------------------------------------------------------------------------
extern "C" void kernel_launch(void* const* d_in, const int* in_sizes, int n_in,
                              void* d_out, int out_size, void* d_ws, size_t ws_size,
                              hipStream_t stream) {
  (void)in_sizes; (void)n_in; (void)out_size; (void)ws_size;

  const float* q       = (const float*)d_in[0];
  const int*   mask    = (const int*)  d_in[3];
  const float* q_gamma = (const float*)d_in[5];
  const float* q_beta  = (const float*)d_in[6];
  const float* q_mean  = (const float*)d_in[7];
  const float* q_var   = (const float*)d_in[8];
  const float* Wq      = (const float*)d_in[9];
  const float* bq      = (const float*)d_in[10];
  const float* Wqq     = (const float*)d_in[11];
  const float* bqq     = (const float*)d_in[12];
  const float* Wo      = (const float*)d_in[13];
  const float* bo      = (const float*)d_in[14];

  float* out  = (float*)d_out;                       // [B,S,D]
  float* attn = out + (size_t)ROWS * DIM;            // [B,H,S,S] region (also logits scratch)

  // Workspace carve-up (≈48 MB).
  char* ws = (char*)d_ws;
  auto carve = [&](size_t bytes) {
    char* p = ws;
    ws += (bytes + 255) & ~(size_t)255;
    return p;
  };
  bf16_t* xb     = (bf16_t*)carve((size_t)ROWS * DIM * 2);
  bf16_t* h1     = (bf16_t*)carve((size_t)ROWS * DIM * 2);
  bf16_t* wqT    = (bf16_t*)carve((size_t)DIM * DIM * 2);
  bf16_t* wqqT   = (bf16_t*)carve((size_t)DIM * DIM * 2);
  bf16_t* woT    = (bf16_t*)carve((size_t)DIM * DIM * 2);
  bf16_t* qh_b   = (bf16_t*)carve((size_t)BH * SEQ * DH * 2);
  bf16_t* qh_t   = (bf16_t*)carve((size_t)BH * DH * SEQ * 2);
  bf16_t* outb   = (bf16_t*)carve((size_t)ROWS * DIM * 2);
  float*  rowmax = (float*)carve((size_t)BH * SEQ * 4);
  float*  rowsum = (float*)carve((size_t)BH * SEQ * 4);
  float*  rowmean= (float*)carve((size_t)BH * SEQ * 4);
  float*  w_with = (float*)carve((size_t)BH * SEQ * 4);
  float*  last2  = (float*)carve((size_t)BH * 4);
  float*  w_betw = (float*)carve((size_t)BH * 4);

  // 1) BN -> bf16
  hipLaunchKernelGGL(k_bn, dim3(ROWS * DIM / 256), dim3(256), 0, stream,
                     q, q_gamma, q_beta, q_mean, q_var, xb);
  // 2) weight transposes -> bf16
  hipLaunchKernelGGL(k_wT, dim3(DIM * DIM / 256), dim3(256), 0, stream, Wq, wqT);
  hipLaunchKernelGGL(k_wT, dim3(DIM * DIM / 256), dim3(256), 0, stream, Wqq, wqqT);
  hipLaunchKernelGGL(k_wT, dim3(DIM * DIM / 256), dim3(256), 0, stream, Wo, woT);
  // 3) h1 = relu(xb @ Wq + bq)
  hipLaunchKernelGGL((k_gemm<0>), dim3(ROWS / 128, DIM / 64), dim3(256), 0, stream,
                     xb, wqT, bq, h1, (float*)nullptr, (bf16_t*)nullptr, ROWS, DIM, DIM);
  // 4) qh / qh^T = split_heads(h1 @ Wqq + bqq)
  hipLaunchKernelGGL((k_gemm<1>), dim3(ROWS / 128, DIM / 64), dim3(256), 0, stream,
                     h1, wqqT, bqq, qh_b, (float*)nullptr, qh_t, ROWS, DIM, DIM);
  // 5) logits (into attn region of d_out)
  hipLaunchKernelGGL(k_logits, dim3(SEQ / 128, SEQ / 64, BH), dim3(256), 0, stream,
                     qh_b, mask, attn);
  // 6) per-row stats (one pass)
  hipLaunchKernelGGL(k_rowstats, dim3(BH * SEQ), dim3(256), 0, stream,
                     attn, rowmax, rowsum, rowmean);
  // 7) per-(b,h) query weights + last2
  hipLaunchKernelGGL(k_within, dim3(BH), dim3(256), 0, stream, rowmean, w_with, last2);
  // 8) per-batch head weights
  hipLaunchKernelGGL(k_betw, dim3(1), dim3(32), 0, stream, last2, w_betw);
  // 9) fused: finalize attn in place + outb = merge_heads(attn @ qh) as bf16 [B,S,D]
  hipLaunchKernelGGL(k_attn_out_fused, dim3(SEQ / 128, 1, BH), dim3(256), 0, stream,
                     attn, qh_t, rowmax, rowsum, w_with, w_betw, outb);
  // 10) out = outb @ Wo + bo (fp32 into d_out)
  hipLaunchKernelGGL((k_gemm<2>), dim3(ROWS / 128, DIM / 64), dim3(256), 0, stream,
                     outb, woT, bo, (bf16_t*)nullptr, out, (bf16_t*)nullptr, ROWS, DIM, DIM);
}
